// InnerPatchAttention_39221641347403
// MI455X (gfx1250) — matrix operand order repaired and checked by reading, not compiled, not run on hardware
//
#include <hip/hip_runtime.h>
#include <hip/hip_bf16.h>
#include <hip/hip_fp16.h>

// ---------------- CDNA5 WMMA plumbing (wave32, 16x16x32 f16 -> f32) --------
typedef __attribute__((ext_vector_type(16))) _Float16 v16h;
typedef __attribute__((ext_vector_type(8)))  float    v8f;

struct alignas(16) U4 { unsigned int x, y, z, w; };   // POD 16-byte chunk

union AFrag {
    v16h     v;
    _Float16 h[16];
    U4       q[2];
};
union P8  { _Float16 h[8];  U4 q;    };               // 8 halves  = 16B
union P16 { _Float16 h[16]; U4 q[2]; };               // 16 halves = 32B

// A operand: row-major M x K (ld = row stride in halves), tile origin `base`.
// ISA layout: lane = M + 16*hh,  element e -> K = (e/8)*16 + hh*8 + (e%8)
__device__ __forceinline__ v16h load_a(const _Float16* base, int ld) {
    int lane = threadIdx.x & 31;
    int m  = lane & 15;
    int hh = lane >> 4;
    const _Float16* p = base + m * ld + hh * 8;
    AFrag f;
    f.q[0] = *reinterpret_cast<const U4*>(p);
    f.q[1] = *reinterpret_cast<const U4*>(p + 16);
    return f.v;
}

// B operand for D = A * W^T (B[k][n] = W[n][k]), W row-major (ldw halves),
// Wrow0 points at W[n0][k0].  ISA layout: lane = N + 16*hh, element e -> K = (e/8)*16 + hh*8 + (e%8)
__device__ __forceinline__ v16h load_b(const _Float16* Wrow0, int ldw) {
    int lane = threadIdx.x & 31;
    int n  = lane & 15;
    int hh = lane >> 4;
    const _Float16* p = Wrow0 + n * ldw + hh * 8;
    AFrag f;
    f.q[0] = *reinterpret_cast<const U4*>(p);
    f.q[1] = *reinterpret_cast<const U4*>(p + 16);
    return f.v;
}

__device__ __forceinline__ v8f wmma16(v16h a, v16h b, v8f c) {
    return __builtin_amdgcn_wmma_f32_16x16x32_f16(false, a, false, b,
                                                  (short)0, c, false, false);
}

// ---------------- constants ------------------------------------------------
#define Bv   2
#define Cv   128
#define Hv   256
#define Pv   16
#define Ev   256
#define NHv  8
#define DHv  32
#define Tv   256          /* tokens per patch */
#define BNv  512          /* B * NP */

// ---------------- kernel 0a: f32 -> f16 weight convert ---------------------
__global__ void k_cvt(const float* __restrict__ a, _Float16* __restrict__ o, int n) {
    int i = blockIdx.x * 256 + threadIdx.x;
    if (i < n) o[i] = (_Float16)a[i];
}

// ---------------- kernel 0b: bp[t][e] = b_pe[e] + pos[t][e] (f16) ----------
__global__ void k_biaspos(const float* __restrict__ bpe, const float* __restrict__ pos,
                          _Float16* __restrict__ bp) {
    int i = blockIdx.x * 256 + threadIdx.x;          // 65536 total
    bp[i] = (_Float16)(pos[i] + bpe[i & 255]);
}

// ---------------- kernel 1: patch embed ------------------------------------
// seq[bn, t, e] = sum_cc z[bn,t,cc] * w_pe[e,cc]  (+ bp[t][e] on copy-out)
// z[bn, 2c + p/8, (p%8)*16 + q] = x[b, c, i*16+p, j*16+q]
__global__ void k_patch_embed(const float* __restrict__ x,
                              const _Float16* __restrict__ wpe,   // [256][128] f16
                              const _Float16* __restrict__ bp,    // [256][256] f16
                              _Float16* __restrict__ seq) {
    extern __shared__ _Float16 sh1[];
    _Float16* zlds = sh1;              // [256][128] = 64 KB
    _Float16* stg  = sh1 + 32768;      // [256][256] = 128 KB
    const int blk = blockIdx.x;        // bn = b*256 + np
    const int b  = blk >> 8, np = blk & 255;
    const int pi = np >> 4,  pj = np & 15;
    const int tid = threadIdx.x;

    // stage z: 2048 contiguous 16-float rows of x -> 32B LDS stores
    #pragma unroll
    for (int sgi = 0; sgi < 8; ++sgi) {
        int seg = tid + sgi * 256;
        int c = seg >> 4, p = seg & 15;
        const float* src = x + (((size_t)(b * Cv + c) * Hv) + pi * Pv + p) * Hv + pj * Pv;
        float tmp[16];
        *reinterpret_cast<float4*>(&tmp[0])  = *reinterpret_cast<const float4*>(src);
        *reinterpret_cast<float4*>(&tmp[4])  = *reinterpret_cast<const float4*>(src + 4);
        *reinterpret_cast<float4*>(&tmp[8])  = *reinterpret_cast<const float4*>(src + 8);
        *reinterpret_cast<float4*>(&tmp[12]) = *reinterpret_cast<const float4*>(src + 12);
        P16 pk;
        #pragma unroll
        for (int e = 0; e < 16; ++e) pk.h[e] = (_Float16)tmp[e];
        int t   = 2 * c + (p >> 3);
        int cc0 = (p & 7) * 16;
        _Float16* dst = zlds + t * Cv + cc0;
        *reinterpret_cast<U4*>(dst)     = pk.q[0];
        *reinterpret_cast<U4*>(dst + 8) = pk.q[1];
    }
    __syncthreads();

    const int lane = tid & 31, wave = tid >> 5, h2 = (tid & 31) >> 4;
    for (int mt = wave * 2; mt < wave * 2 + 2; ++mt) {
        v16h a[4];
        #pragma unroll
        for (int k = 0; k < 4; ++k) a[k] = load_a(zlds + mt * 16 * Cv + k * 32, Cv);
        for (int nt = 0; nt < 16; ++nt) {
            v8f acc = {};
            #pragma unroll
            for (int k = 0; k < 4; ++k) {
                v16h bf = load_b(wpe + (nt * 16) * Cv + k * 32, Cv);
                acc = wmma16(a[k], bf, acc);
            }
            int n = nt * 16 + (lane & 15);
            #pragma unroll
            for (int r = 0; r < 8; ++r)
                stg[(mt * 16 + r + 8 * h2) * Ev + n] = (_Float16)acc[r];
        }
    }
    __syncthreads();

    // coalesced copy-out, folding in bp = b_pe + pos (packed f16 adds)
    {
        const U4* srow = reinterpret_cast<const U4*>(stg + tid * Ev);
        const U4* brow = reinterpret_cast<const U4*>(bp + tid * Ev);
        U4* drow = reinterpret_cast<U4*>(seq + ((size_t)blk * Tv + tid) * Ev);
        #pragma unroll 4
        for (int u = 0; u < Ev / 8; ++u) {
            P8 av, bv;
            av.q = srow[u];
            bv.q = brow[u];
            #pragma unroll
            for (int e = 0; e < 8; ++e) av.h[e] = av.h[e] + bv.h[e];
            drow[u] = av.q;
        }
    }
}

// ---------------- kernel 2: packed QKV projection --------------------------
__global__ void k_qkv(const _Float16* __restrict__ seq,
                      const _Float16* __restrict__ win,   // [768][256] f16
                      const float* __restrict__ bin,      // [768]
                      _Float16* __restrict__ qb,          // [BN,NH,T,DH] (pre-scaled)
                      _Float16* __restrict__ kb,          // [BN,NH,T,DH]
                      _Float16* __restrict__ vtb) {       // [BN,NH,DH,T] (transposed)
    extern __shared__ _Float16 sh2[];
    _Float16* s_s = sh2;               // [256][256] seq tile, 128 KB
    _Float16* stg = sh2 + 65536;       // [256][256] Q/K stage, 128 KB
    const int blk = blockIdx.x;        // bn
    const int tid = threadIdx.x, lane = tid & 31, wave = tid >> 5, h2 = (tid & 31) >> 4;
    const float scale = 0.17677669529663687f;            // 1/sqrt(32)

    // async-stage seq tile straight into LDS (GVS: SGPR base + lane offset);
    // LDS byte address = low 32 bits of the generic shared-memory pointer.
    {
        unsigned lbase = (unsigned)(uintptr_t)s_s;
        const char* gbase = (const char*)(seq + (size_t)blk * Tv * Ev);
        for (int i = tid; i < (Tv * Ev) / 8; i += 256) {
            unsigned loff = lbase + (unsigned)i * 16u;
            unsigned goff = (unsigned)i * 16u;
            asm volatile("global_load_async_to_lds_b128 %0, %1, %2"
                         :: "v"(loff), "v"(goff), "s"(gbase) : "memory");
        }
        asm volatile("s_wait_asynccnt 0x0" ::: "memory");
        __syncthreads();
    }

    v8f zero = {};
    for (int g = 0; g < 3; ++g) {                        // 0=Q 1=K 2=V
        for (int mt = wave * 2; mt < wave * 2 + 2; ++mt) {
            for (int sub = 0; sub < 2; ++sub) {
                v8f acc[8];
                #pragma unroll
                for (int i = 0; i < 8; ++i) acc[i] = zero;
                for (int k = 0; k < 8; ++k) {
                    v16h a = load_a(s_s + mt * 16 * Ev + k * 32, Ev);
                    #pragma unroll
                    for (int nt = 0; nt < 8; ++nt) {
                        int n0 = g * 256 + sub * 128 + nt * 16;
                        v16h bf = load_b(win + n0 * Ev + k * 32, Ev);
                        acc[nt] = wmma16(a, bf, acc[nt]);
                    }
                }
                #pragma unroll
                for (int nt = 0; nt < 8; ++nt) {
                    int ncol = sub * 128 + nt * 16 + (lane & 15);   // 0..255
                    float bias = bin[g * 256 + ncol];
                    if (g == 2) {
                        // V: transposed layout -> 8 consecutive halves, one b128 store
                        int head = ncol >> 5, d = ncol & 31;
                        P8 pk;
                        #pragma unroll
                        for (int r = 0; r < 8; ++r)
                            pk.h[r] = (_Float16)(acc[nt][r] + bias);
                        int m0 = mt * 16 + 8 * h2;
                        *reinterpret_cast<U4*>(
                            vtb + ((size_t)(blk * NHv + head) * DHv + d) * Tv + m0) = pk.q;
                    } else {
                        float sc = (g == 0) ? scale : 1.0f;
                        #pragma unroll
                        for (int r = 0; r < 8; ++r)
                            stg[(mt * 16 + r + 8 * h2) * Ev + ncol] =
                                (_Float16)((acc[nt][r] + bias) * sc);
                    }
                }
            }
        }
        if (g < 2) {
            __syncthreads();
            // coalesced copy-out: thread tid owns token row m = tid
            _Float16* dstb = (g == 0) ? qb : kb;
            const U4* srow = reinterpret_cast<const U4*>(stg + tid * Ev);
            #pragma unroll
            for (int head = 0; head < NHv; ++head) {
                U4* drow = reinterpret_cast<U4*>(
                    dstb + ((size_t)(blk * NHv + head) * Tv + tid) * DHv);
                #pragma unroll
                for (int u = 0; u < 4; ++u) drow[u] = srow[head * 4 + u];
            }
            __syncthreads();
        }
    }
}

// ---------------- kernel 3: per (patch, head) attention --------------------
__global__ void k_attn(const _Float16* __restrict__ qb,
                       const _Float16* __restrict__ kb,
                       const _Float16* __restrict__ vtb,
                       _Float16* __restrict__ obuf) {     // [BN,T,E] f16
    extern __shared__ _Float16 slds[];                   // scores [256][256] = 128 KB
    const int blk  = blockIdx.x;                         // bn*8 + head
    const int bn   = blk >> 3, head = blk & 7;
    const _Float16* Q = qb  + (size_t)blk * Tv * DHv;
    const _Float16* K = kb  + (size_t)blk * Tv * DHv;
    const _Float16* V = vtb + (size_t)blk * DHv * Tv;    // [32][256]
    const int tid = threadIdx.x, lane = tid & 31, wave = tid >> 5, h2 = (tid & 31) >> 4;

    // S = Q @ K^T   (single K-chunk of 32 per tile)
    for (int mt = wave * 2; mt < wave * 2 + 2; ++mt) {
        v16h a = load_a(Q + mt * 16 * DHv, DHv);
        for (int nt = 0; nt < 16; ++nt) {
            v16h bf = load_b(K + (nt * 16) * DHv, DHv);
            v8f acc = {};
            acc = wmma16(a, bf, acc);
            #pragma unroll
            for (int r = 0; r < 8; ++r)
                slds[(mt * 16 + r + 8 * h2) * Tv + nt * 16 + (lane & 15)] = (_Float16)acc[r];
        }
    }
    __syncthreads();

    // row-wise softmax, one thread per row
    {
        _Float16* row = slds + tid * Tv;
        float mx = -1e30f;
        for (int n = 0; n < Tv; ++n) mx = fmaxf(mx, (float)row[n]);
        float sum = 0.f;
        for (int n = 0; n < Tv; ++n) {
            float e = __expf((float)row[n] - mx);
            sum += e;
            row[n] = (_Float16)e;
        }
        float inv = 1.0f / sum;
        for (int n = 0; n < Tv; ++n) row[n] = (_Float16)((float)row[n] * inv);
    }
    __syncthreads();

    // O = P @ V ; then park O in cols [0,32) of this wave's own score rows
    v8f zero = {};
    for (int mt = wave * 2; mt < wave * 2 + 2; ++mt) {
        v8f acc[2] = { zero, zero };
        for (int k = 0; k < 8; ++k) {
            v16h a = load_a(slds + mt * 16 * Tv + k * 32, Tv);
            #pragma unroll
            for (int nt = 0; nt < 2; ++nt) {
                v16h bf = load_b(V + (nt * 16) * Tv + k * 32, Tv);
                acc[nt] = wmma16(a, bf, acc[nt]);
            }
        }
        #pragma unroll
        for (int nt = 0; nt < 2; ++nt)
            #pragma unroll
            for (int r = 0; r < 8; ++r)
                slds[(mt * 16 + r + 8 * h2) * Tv + nt * 16 + (lane & 15)] =
                    (_Float16)acc[nt][r];
    }
    __syncthreads();

    // coalesced copy-out: row tid, 64B contiguous into this head's column slab
    {
        const U4* srow = reinterpret_cast<const U4*>(slds + tid * Tv);
        U4* drow = reinterpret_cast<U4*>(obuf + ((size_t)bn * Tv + tid) * Ev + head * DHv);
        #pragma unroll
        for (int u = 0; u < 4; ++u) drow[u] = srow[u];
    }
}

// ---------------- kernel 4: out-proj + LN + Linear(E,1) + sigmoid ----------
__global__ void k_out(const _Float16* __restrict__ obuf,
                      const _Float16* __restrict__ wout,  // [256][256] f16
                      const float* __restrict__ bout,
                      const float* __restrict__ lng, const float* __restrict__ lnb,
                      const float* __restrict__ wlin, const float* __restrict__ blin,
                      float* __restrict__ out) {          // [B,1,256,256]
    extern __shared__ float olds[];                      // [256][256] f32 = 256 KB
    const int blk = blockIdx.x;                          // bn
    const int b  = blk >> 8, np = blk & 255;
    const int pi = np >> 4,  pj = np & 15;
    const _Float16* O = obuf + (size_t)blk * Tv * Ev;
    const int tid = threadIdx.x, lane = tid & 31, wave = tid >> 5, h2 = (tid & 31) >> 4;

    v8f zero = {};
    for (int mt = wave * 2; mt < wave * 2 + 2; ++mt) {
        for (int sub = 0; sub < 2; ++sub) {
            v8f acc[8];
            #pragma unroll
            for (int i = 0; i < 8; ++i) acc[i] = zero;
            for (int k = 0; k < 8; ++k) {
                v16h a = load_a(O + mt * 16 * Ev + k * 32, Ev);
                #pragma unroll
                for (int nt = 0; nt < 8; ++nt) {
                    int n0 = sub * 128 + nt * 16;
                    v16h bf = load_b(wout + n0 * Ev + k * 32, Ev);
                    acc[nt] = wmma16(a, bf, acc[nt]);
                }
            }
            #pragma unroll
            for (int nt = 0; nt < 8; ++nt) {
                int n = sub * 128 + nt * 16 + (lane & 15);
                float bias = bout[n];
                #pragma unroll
                for (int r = 0; r < 8; ++r) {
                    int m = mt * 16 + r + 8 * h2;
                    olds[m * Ev + n] = acc[nt][r] + bias;
                }
            }
        }
    }
    __syncthreads();

    // one thread per row (token t = tid): LN -> dot(w_lin) -> sigmoid -> scatter
    {
        float* row = olds + tid * Ev;
        float s = 0.f, s2 = 0.f;
        for (int n = 0; n < Ev; ++n) { float v = row[n]; s += v; s2 += v * v; }
        float mu  = s * (1.0f / Ev);
        float var = s2 * (1.0f / Ev) - mu * mu;
        float rstd = rsqrtf(var + 1e-5f);
        float dot = 0.f;
        for (int n = 0; n < Ev; ++n) {
            float ln = (row[n] - mu) * rstd * lng[n] + lnb[n];
            dot += ln * wlin[n];
        }
        float z   = dot + blin[0];
        float sig = 1.0f / (1.0f + __expf(-z));
        int p = tid >> 4, q = tid & 15;
        out[((size_t)b * Hv + pi * Pv + p) * Hv + pj * Pv + q] = sig;
    }
}

// ---------------- launch ---------------------------------------------------
extern "C" void kernel_launch(void* const* d_in, const int* in_sizes, int n_in,
                              void* d_out, int out_size, void* d_ws, size_t ws_size,
                              hipStream_t stream) {
    (void)in_sizes; (void)n_in; (void)out_size; (void)ws_size;
    const float* x    = (const float*)d_in[0];
    const float* wpe  = (const float*)d_in[1];
    const float* bpe  = (const float*)d_in[2];
    const float* pos  = (const float*)d_in[3];
    const float* win  = (const float*)d_in[4];
    const float* bin  = (const float*)d_in[5];
    const float* wout = (const float*)d_in[6];
    const float* bout = (const float*)d_in[7];
    const float* lng  = (const float*)d_in[8];
    const float* lnb  = (const float*)d_in[9];
    const float* wlin = (const float*)d_in[10];
    const float* blin = (const float*)d_in[11];
    float* out = (float*)d_out;

    // workspace layout (halves)
    _Float16* ws     = (_Float16*)d_ws;
    _Float16* wpe_h  = ws;                                  //  32768
    _Float16* win_h  = ws + 32768;                          // 196608
    _Float16* wout_h = ws + 229376;                         //  65536
    _Float16* bp     = ws + 294912;                         //  65536 (b_pe + pos, f16)
    _Float16* seq    = ws + 360448;                         // 33554432 (reused as obuf)
    _Float16* qb     = seq + 33554432ull;
    _Float16* kb     = qb  + 33554432ull;
    _Float16* vtb    = kb  + 33554432ull;
    _Float16* ob     = seq;                                 // alias: seq dead after k_qkv

    hipFuncSetAttribute((const void*)k_patch_embed, hipFuncAttributeMaxDynamicSharedMemorySize, 192 * 1024);
    hipFuncSetAttribute((const void*)k_qkv,         hipFuncAttributeMaxDynamicSharedMemorySize, 256 * 1024);
    hipFuncSetAttribute((const void*)k_attn,        hipFuncAttributeMaxDynamicSharedMemorySize, 128 * 1024);
    hipFuncSetAttribute((const void*)k_out,         hipFuncAttributeMaxDynamicSharedMemorySize, 256 * 1024);

    k_cvt<<<(32768  + 255) / 256, 256, 0, stream>>>(wpe,  wpe_h,  32768);
    k_cvt<<<(196608 + 255) / 256, 256, 0, stream>>>(win,  win_h,  196608);
    k_cvt<<<(65536  + 255) / 256, 256, 0, stream>>>(wout, wout_h, 65536);
    k_biaspos<<<256, 256, 0, stream>>>(bpe, pos, bp);

    k_patch_embed<<<BNv, 256, 192 * 1024, stream>>>(x, wpe_h, bp, seq);
    k_qkv        <<<BNv, 256, 256 * 1024, stream>>>(seq, win_h, bin, qb, kb, vtb);
    k_attn       <<<BNv * NHv, 256, 128 * 1024, stream>>>(qb, kb, vtb, ob);
    k_out        <<<BNv, 256, 256 * 1024, stream>>>(ob, wout_h, bout, lng, lnb, wlin, blin, out);
}
